// DoomLiquidNet_8847632630113
// MI455X (gfx1250) — matrix operand order repaired and checked by reading, not compiled, not run on hardware
//
#include <hip/hip_runtime.h>
#include <hip/hip_bf16.h>

typedef __attribute__((ext_vector_type(16))) _Float16 v16h;
typedef __attribute__((ext_vector_type(8)))  float    v8f;
typedef __attribute__((ext_vector_type(4)))  unsigned v4u;

union FragU { unsigned u[8]; v4u q[2]; v16h h; };

__device__ __forceinline__ unsigned packh(float a, float b) {
    union { _Float16 h[2]; unsigned u; } r;
    r.h[0] = (_Float16)a; r.h[1] = (_Float16)b; return r.u;
}

// 16-bit A-matrix 16x32 per-lane K offsets (ISA 7.12.2): lane half = lane>>4.
__device__ __forceinline__ int a_koff(int v, int half) {
    return ((v < 4) ? 0 : 16) + half * 8 + (v & 3) * 2;
}
// 16-bit B-matrix 32x16 per-lane K offsets (same K order as A).
__device__ __forceinline__ int b_koff(int v, int half) {
    return ((v < 4) ? 0 : 16) + half * 8 + (v & 3) * 2;
}

__device__ __forceinline__ v8f wmma_f16(const FragU& a, const FragU& b, v8f c) {
    return __builtin_amdgcn_wmma_f32_16x16x32_f16(
        /*neg_a=*/false, a.h, /*neg_b=*/false, b.h,
        /*c_mod=*/(short)0, c, /*reuse_a=*/false, /*reuse_b=*/false);
}

// ---------------------------------------------------------------------------
// Problem constants
// ---------------------------------------------------------------------------
#define IMGS   2048        // B*T
#define C1_K   48          // 3*4*4  (padded to 64 -> 2 ksteps)
#define C1_SP  961         // 31*31
#define C1_NT  61          // ceil(961/16)
#define C2_K   512         // 32*4*4 (16 ksteps)
#define C2_SP  196         // 14*14
#define C2_NT  13          // ceil(196/16)
#define FLATK  12544       // 64*196
#define ZK     12608       // FLAT + UNITS
#define BKS    394         // 12608/32
#define BKS_G  392         // ksteps fully inside the FLAT (global) region
// packed sizes (dwords)
#define N_W1A  (4   * 256)         // 2 mt * 2 ks
#define N_W2A  (64  * 256)         // 4 mt * 16 ks
#define N_BBB  (3152 * 256)        // 8 nt * 394 ks
#define N_HB   (16  * 256)         // 4 nt * 4 ks (per head matrix)

// ---------------------------------------------------------------------------
// Weight prep: fp32 -> f16, packed into exact WMMA per-lane fragment order.
// ---------------------------------------------------------------------------
__global__ void prep_kernel(const float* __restrict__ w1, const float* __restrict__ w2,
                            const float* __restrict__ bbw,
                            const float* __restrict__ f1w, const float* __restrict__ f2w,
                            const float* __restrict__ taw, const float* __restrict__ tbw,
                            const float* __restrict__ tab, const float* __restrict__ tbb,
                            unsigned* __restrict__ w1p, unsigned* __restrict__ w2p,
                            unsigned* __restrict__ bbp, unsigned* __restrict__ headp,
                            float* __restrict__ tbs)
{
    const long total = (long)N_W1A + N_W2A + N_BBB + 3L * N_HB + 64;
    for (long idx = (long)blockIdx.x * blockDim.x + threadIdx.x; idx < total;
         idx += (long)gridDim.x * blockDim.x) {
        long i = idx;
        if (i < N_W1A) {                       // conv1 A-pack: M=32, K=48 (pad 64)
            int tile = (int)(i >> 8), lane = (int)((i >> 3) & 31), v = (int)(i & 7);
            int mt = tile >> 1, ks = tile & 1;
            int m = mt * 16 + (lane & 15);
            int k = ks * 32 + a_koff(v, lane >> 4);
            float x0 = (k     < C1_K) ? w1[m * C1_K + k]     : 0.f;
            float x1 = (k + 1 < C1_K) ? w1[m * C1_K + k + 1] : 0.f;
            w1p[i] = packh(x0, x1);
            continue;
        }
        i -= N_W1A;
        if (i < N_W2A) {                       // conv2 A-pack: M=64, K=512
            int tile = (int)(i >> 8), lane = (int)((i >> 3) & 31), v = (int)(i & 7);
            int mt = tile >> 4, ks = tile & 15;
            int m = mt * 16 + (lane & 15);
            int k = ks * 32 + a_koff(v, lane >> 4);
            w2p[i] = packh(w2[m * C2_K + k], w2[m * C2_K + k + 1]);
            continue;
        }
        i -= N_W2A;
        if (i < N_BBB) {                       // backbone B-pack: B(k,n)=bb_w[n][k]
            int tile = (int)(i >> 8), lane = (int)((i >> 3) & 31), v = (int)(i & 7);
            int nt = tile / BKS, ks = tile % BKS;
            int n = nt * 16 + (lane & 15);
            long k = (long)ks * 32 + b_koff(v, lane >> 4);
            bbp[i] = packh(bbw[(long)n * ZK + k], bbw[(long)n * ZK + k + 1]);
            continue;
        }
        i -= N_BBB;
        if (i < 3L * N_HB) {                   // head B-pack: ff1, ff2, (ta+tb)
            int q = (int)(i / N_HB); long j = i % N_HB;
            int tile = (int)(j >> 8), lane = (int)((j >> 3) & 31), v = (int)(j & 7);
            int nt = tile >> 2, ks = tile & 3;
            int n = nt * 16 + (lane & 15);
            int k = ks * 32 + b_koff(v, lane >> 4);
            float s0, s1;
            if (q == 0)      { s0 = f1w[n * 128 + k]; s1 = f1w[n * 128 + k + 1]; }
            else if (q == 1) { s0 = f2w[n * 128 + k]; s1 = f2w[n * 128 + k + 1]; }
            else { s0 = taw[n * 128 + k] + tbw[n * 128 + k];
                   s1 = taw[n * 128 + k + 1] + tbw[n * 128 + k + 1]; }
            headp[i] = packh(s0, s1);
            continue;
        }
        i -= 3L * N_HB;                        // fused time-gate bias
        tbs[i] = tab[i] + tbb[i];
    }
}

// ---------------------------------------------------------------------------
// Conv1 implicit GEMM: per wave, D[32,16] = W[32,48] x patch[48,16].
// Output f16, layout [img][oc][31][32] (row padded to 32 for aligned u32 reads).
// All loads are unconditional (clamped addresses); predication only via
// v_cndmask selects and masked stores -- no divergent branches around loads.
// ---------------------------------------------------------------------------
__global__ __launch_bounds__(32) void conv1_kernel(const float* __restrict__ x,
        const float* __restrict__ b1, const unsigned* __restrict__ w1p,
        _Float16* __restrict__ c1out)
{
    int lane = threadIdx.x, nloc = lane & 15, half = lane >> 4;
    int ntile = blockIdx.x, img = blockIdx.y;
    int s = ntile * 16 + nloc;
    bool valid = s < C1_SP;
    int ss = valid ? s : 0;
    int oh = ss / 31, ow = ss % 31;
    const float* xin = x + (size_t)img * (3 * 64 * 64);
    const float* xpix = xin + (oh * 2) * 64 + (ow * 2);   // top-left of this patch

    FragU a[2][2];
#pragma unroll
    for (int mt = 0; mt < 2; ++mt)
#pragma unroll
        for (int ks = 0; ks < 2; ++ks) {
            const unsigned* wp = w1p + (size_t)((mt * 2 + ks) * 256 + lane * 8);
            a[mt][ks].q[0] = *(const v4u*)(wp);
            a[mt][ks].q[1] = *(const v4u*)(wp + 4);
        }

    v8f zero = {0.f,0.f,0.f,0.f,0.f,0.f,0.f,0.f};
    v8f c0 = zero, c1 = zero;

    // ---- kstep 0: k = b_koff(v,half) in [0,30], always < 48 -> unconditional ----
    {
        FragU bf;
#pragma unroll
        for (int v = 0; v < 8; ++v) {
            int k = b_koff(v, half);
            int ic = k >> 4, kr = (k >> 2) & 3, kc = k & 3;       // kc even
            const float* p = xpix + ic * 4096 + kr * 64 + kc;
            bf.u[v] = packh(p[0], p[1]);
        }
        c0 = wmma_f16(a[0][0], bf, c0);
        c1 = wmma_f16(a[1][0], bf, c1);
    }
    // ---- kstep 1: real k = 32 + b_koff(v,half); valid only for v<4.
    // All lanes load k = 32 + half*8 + 2(v&3) (always in-bounds), v>=4 selects 0
    // (the packed A fragment is also zero there, so this is doubly safe). ----
    {
        FragU bf;
#pragma unroll
        for (int v = 0; v < 8; ++v) {
            int k = 32 + half * 8 + (v & 3) * 2;                   // 32..46
            int ic = k >> 4, kr = (k >> 2) & 3, kc = k & 3;
            const float* p = xpix + ic * 4096 + kr * 64 + kc;
            unsigned val = packh(p[0], p[1]);
            bf.u[v] = (v < 4) ? val : 0u;                          // v_cndmask
        }
        c0 = wmma_f16(a[0][1], bf, c0);
        c1 = wmma_f16(a[1][1], bf, c1);
    }

    if (valid) {
#pragma unroll
        for (int r = 0; r < 8; ++r) {
            int oc0 = half * 8 + r;
            float v0 = c0[r] + b1[oc0]; v0 = v0 > 0.f ? v0 : 0.f;
            c1out[((size_t)(img * 32 + oc0) * 31 + oh) * 32 + ow] = (_Float16)v0;
            int oc1 = 16 + oc0;
            float v1 = c1[r] + b1[oc1]; v1 = v1 > 0.f ? v1 : 0.f;
            c1out[((size_t)(img * 32 + oc1) * 31 + oh) * 32 + ow] = (_Float16)v1;
        }
    }
}

// ---------------------------------------------------------------------------
// Conv2 implicit GEMM: per wave, D[64,16] = W[64,512] x patch[512,16].
// Output f16 r_in layout [img][oc*196+s] == [img][FLAT].
// ---------------------------------------------------------------------------
__global__ __launch_bounds__(32) void conv2_kernel(const _Float16* __restrict__ c1out,
        const float* __restrict__ b2, const unsigned* __restrict__ w2p,
        _Float16* __restrict__ c2out)
{
    int lane = threadIdx.x, nloc = lane & 15, half = lane >> 4;
    int ntile = blockIdx.x, img = blockIdx.y;
    int s = ntile * 16 + nloc;
    bool valid = s < C2_SP;
    int ss = valid ? s : 0;
    int oy = ss / 14, ox = ss % 14;
    const _Float16* zin = c1out + (size_t)img * 32 * 31 * 32
                        + (size_t)(oy * 2) * 32 + ox * 2;          // patch top-left

    v8f zero = {0.f,0.f,0.f,0.f,0.f,0.f,0.f,0.f};
    v8f c[4] = {zero, zero, zero, zero};
    for (int ks = 0; ks < 16; ++ks) {
        FragU bf;
#pragma unroll
        for (int v = 0; v < 8; ++v) {
            int k = ks * 32 + b_koff(v, half);                     // always < 512
            int ic = k >> 4, kr = (k >> 2) & 3, kc = k & 3;        // kc even
            bf.u[v] = *(const unsigned*)(zin + (size_t)(ic * 31 + kr) * 32 + kc);
        }
#pragma unroll
        for (int mt = 0; mt < 4; ++mt) {
            FragU af;
            const unsigned* wp = w2p + (size_t)((mt * 16 + ks) * 256 + lane * 8);
            af.q[0] = *(const v4u*)(wp);
            af.q[1] = *(const v4u*)(wp + 4);
            c[mt] = wmma_f16(af, bf, c[mt]);
        }
    }
    if (valid) {
#pragma unroll
        for (int mt = 0; mt < 4; ++mt)
#pragma unroll
            for (int r = 0; r < 8; ++r) {
                int oc = mt * 16 + half * 8 + r;
                float v = c[mt][r] + b2[oc]; v = v > 0.f ? v : 0.f;
                c2out[(size_t)(img * 64 + oc) * C2_SP + ss] = (_Float16)v;
            }
    }
}

// ---------------------------------------------------------------------------
// CfC recurrence: 1 workgroup, 16 waves. Per t:
//   phase A: wave (mt,nt) computes backbone tile [16x16] over K=12608:
//            ks 0..391 from global r_in (2 x b128 per fragment), ks 392..393
//            from the f16 hidden state in LDS.
//   phase B: 24 head tile-jobs (ff1/ff2/tgate, K=128) from LDS-staged bb
//   phase C: fp32 tanh/sigmoid combine -> h (LDS f16 + global fp32)
// ---------------------------------------------------------------------------
__global__ __launch_bounds__(512) void cfc_kernel(const _Float16* __restrict__ r_in,
        const float* __restrict__ bbb, const unsigned* __restrict__ bbp,
        const unsigned* __restrict__ headp,
        const float* __restrict__ f1b, const float* __restrict__ f2b,
        const float* __restrict__ tbs, float* __restrict__ h_out)
{
    __shared__ _Float16 bbL[32][128];       // backbone activations (f16)
    __shared__ float    headL[3][32][64];   // raw head outputs (fp32)
    __shared__ _Float16 hbuf[32][64];       // hidden state (f16, feeds next zz)

    int tid = threadIdx.x, wave = tid >> 5, lane = tid & 31;
    int nloc = lane & 15, half = lane >> 4;

    for (int i = tid; i < 32 * 64; i += 512) (&hbuf[0][0])[i] = (_Float16)0.f;
    __syncthreads();

    int mt = wave >> 3, nt = wave & 7;          // backbone tile of this wave
    int m  = mt * 16 + nloc;                    // batch row held by this lane (A)
    v8f zero = {0.f,0.f,0.f,0.f,0.f,0.f,0.f,0.f};

    for (int t = 0; t < 64; ++t) {
        // ---- phase A: backbone GEMM tile ----
        v8f acc = zero;
        const _Float16* zrow = r_in + (size_t)(m * 64 + t) * FLATK + half * 8;
        const unsigned* bprow = bbp + ((size_t)nt * BKS * 256 + lane * 8);
        for (int ks = 0; ks < BKS_G; ++ks) {
            if ((ks & 7) == 0)
                __builtin_prefetch(zrow + ks * 32 + 1024, 0, 0);
            FragU af, bf;
            af.q[0] = *(const v4u*)(zrow + ks * 32);        // k = ks*32 + half*8 + 0..7
            af.q[1] = *(const v4u*)(zrow + ks * 32 + 16);   // k = ks*32 + 16 + half*8 ..
            const unsigned* bp = bprow + (size_t)ks * 256;
            bf.q[0] = *(const v4u*)(bp);
            bf.q[1] = *(const v4u*)(bp + 4);
            acc = wmma_f16(af, bf, acc);
        }
#pragma unroll
        for (int ks = BKS_G; ks < BKS; ++ks) {              // hidden-state tail (LDS)
            FragU af, bf;
            const _Float16* hp = &hbuf[m][0] + (ks - BKS_G) * 32 + half * 8;
            af.q[0] = *(const v4u*)(hp);
            af.q[1] = *(const v4u*)(hp + 16);
            const unsigned* bp = bprow + (size_t)ks * 256;
            bf.q[0] = *(const v4u*)(bp);
            bf.q[1] = *(const v4u*)(bp + 4);
            acc = wmma_f16(af, bf, acc);
        }
        {
            int j = nt * 16 + nloc;
#pragma unroll
            for (int r = 0; r < 8; ++r) {
                int b = mt * 16 + half * 8 + r;
                float v = acc[r] + bbb[j];
                bbL[b][j] = (_Float16)(1.7159f * tanhf(0.666f * v));
            }
        }
        __syncthreads();

        // ---- phase B: head GEMMs (wave-uniform job loop; EXEC stays full) ----
        for (int job = wave; job < 24; job += 16) {
            int q = job >> 3, rem = job & 7, hmt = rem >> 2, hnt = rem & 3;
            v8f hc = zero;
            int hm = hmt * 16 + nloc;
#pragma unroll
            for (int ks = 0; ks < 4; ++ks) {
                FragU af, bf;
                const _Float16* ap = &bbL[hm][0] + ks * 32 + half * 8;
                af.q[0] = *(const v4u*)(ap);
                af.q[1] = *(const v4u*)(ap + 16);
                const unsigned* bp = headp + ((size_t)q * N_HB
                                              + (hnt * 4 + ks) * 256 + lane * 8);
                bf.q[0] = *(const v4u*)(bp);
                bf.q[1] = *(const v4u*)(bp + 4);
                hc = wmma_f16(af, bf, hc);
            }
            int u = hnt * 16 + nloc;
            float bias = (q == 0) ? f1b[u] : (q == 1) ? f2b[u] : tbs[u];
#pragma unroll
            for (int r = 0; r < 8; ++r) {
                int b = hmt * 16 + half * 8 + r;
                headL[q][b][u] = hc[r] + bias;
            }
        }
        __syncthreads();

        // ---- phase C: CfC combine ----
        for (int i = tid; i < 2048; i += 512) {
            int b = i >> 6, u = i & 63;
            float f1 = tanhf(headL[0][b][u]);
            float f2 = tanhf(headL[1][b][u]);
            float ti = 1.f / (1.f + expf(-headL[2][b][u]));
            float h  = f1 * (1.f - ti) + ti * f2;
            hbuf[b][u] = (_Float16)h;
            h_out[(size_t)(b * 64 + t) * 64 + u] = h;   // [b][t][u]
        }
        __syncthreads();
    }
}

// ---------------------------------------------------------------------------
// Output head (fp32): logits[b,t,:] = h @ out_w.T + out_b ; hT copy.
// ---------------------------------------------------------------------------
__global__ void logits_kernel(const float* __restrict__ h_out,
                              const float* __restrict__ ow, const float* __restrict__ ob,
                              float* __restrict__ out)
{
    int i = blockIdx.x * blockDim.x + threadIdx.x;   // b*64 + t
    if (i >= 2048) return;
    const float* h = h_out + (size_t)i * 64;
#pragma unroll
    for (int a = 0; a < 8; ++a) {
        float acc = ob[a];
        for (int k = 0; k < 64; ++k) acc += h[k] * ow[a * 64 + k];
        out[(size_t)i * 8 + a] = acc;
    }
    if ((i & 63) == 63) {
        int b = i >> 6;
        for (int u = 0; u < 64; ++u) out[16384 + b * 64 + u] = h[u];
    }
}

// ---------------------------------------------------------------------------
extern "C" void kernel_launch(void* const* d_in, const int* in_sizes, int n_in,
                              void* d_out, int out_size, void* d_ws, size_t ws_size,
                              hipStream_t stream)
{
    (void)in_sizes; (void)n_in; (void)out_size; (void)ws_size;
    const float* x   = (const float*)d_in[0];
    const float* w1  = (const float*)d_in[1];
    const float* b1  = (const float*)d_in[2];
    const float* w2  = (const float*)d_in[3];
    const float* b2  = (const float*)d_in[4];
    const float* bbw = (const float*)d_in[5];
    const float* bbb = (const float*)d_in[6];
    const float* f1w = (const float*)d_in[7];
    const float* f1b = (const float*)d_in[8];
    const float* f2w = (const float*)d_in[9];
    const float* f2b = (const float*)d_in[10];
    const float* taw = (const float*)d_in[11];
    const float* tab = (const float*)d_in[12];
    const float* tbw = (const float*)d_in[13];
    const float* tbb = (const float*)d_in[14];
    const float* ow  = (const float*)d_in[15];
    const float* ob  = (const float*)d_in[16];
    float* out = (float*)d_out;

    char* ws = (char*)d_ws;
    size_t cur = 0;
    auto alloc = [&](size_t bytes) -> char* {
        char* p = ws + cur;
        cur = (cur + bytes + 255) & ~(size_t)255;
        return p;
    };
    _Float16* c1out = (_Float16*)alloc((size_t)IMGS * 32 * 31 * 32 * 2); // padded rows
    _Float16* c2out = (_Float16*)alloc((size_t)IMGS * FLATK * 2);
    float*    h_out = (float*)   alloc((size_t)32 * 64 * 64 * 4);
    unsigned* w1p   = (unsigned*)alloc((size_t)N_W1A * 4);
    unsigned* w2p   = (unsigned*)alloc((size_t)N_W2A * 4);
    unsigned* bbp   = (unsigned*)alloc((size_t)N_BBB * 4);
    unsigned* headp = (unsigned*)alloc((size_t)3 * N_HB * 4);
    float*    tbs   = (float*)   alloc(64 * 4);

    long prep_total = (long)N_W1A + N_W2A + N_BBB + 3L * N_HB + 64;
    prep_kernel<<<(unsigned)((prep_total + 255) / 256), 256, 0, stream>>>(
        w1, w2, bbw, f1w, f2w, taw, tbw, tab, tbb, w1p, w2p, bbp, headp, tbs);
    conv1_kernel<<<dim3(C1_NT, IMGS), 32, 0, stream>>>(x, b1, w1p, c1out);
    conv2_kernel<<<dim3(C2_NT, IMGS), 32, 0, stream>>>(c1out, b2, w2p, c2out);
    cfc_kernel<<<1, 512, 0, stream>>>(c2out, bbb, bbp, headp, f1b, f2b, tbs, h_out);
    logits_kernel<<<8, 256, 0, stream>>>(h_out, ow, ob, out);
}
